// MultiHeadAttention_919123001362
// MI455X (gfx1250) — compile-verified, matrix operand order checked by reading, not run on hardware
//
#include <hip/hip_runtime.h>
#include <math.h>

// ---- problem constants -----------------------------------------------------
constexpr int kB  = 64;
constexpr int kN  = 577;
constexpr int kE  = 768;
constexpr int kH  = 8;
constexpr int kD  = 96;               // head dim
constexpr int kNP = 608;              // N padded to multiple of 32 (19*32)
constexpr int kBN = kB * kN;          // 36928 rows (multiple of 16: 2308*16)
constexpr float kScale = 0.1020620726159658f;  // D^-0.5 = 1/sqrt(96)

typedef __attribute__((ext_vector_type(16))) _Float16 v16h;
typedef __attribute__((ext_vector_type(8)))  float    v8f;

__device__ __forceinline__ v8f wmma_f16(v16h a, v16h b, v8f c) {
  return __builtin_amdgcn_wmma_f32_16x16x32_f16(false, a, false, b,
                                                (short)0, c, false, false);
}

// Load a 16x32 f16 fragment (A layout; B layout is the mirrored case with the
// same lane pattern: lane%16 selects row/col, K split {0..7,16..23} for lanes
// 0-15 and {8..15,24..31} for lanes 16-31). Contiguous along K -> 2x b128.
__device__ __forceinline__ v16h load_frag16(const _Float16* __restrict__ base,
                                            int stride, int lane) {
  const _Float16* p = base + (size_t)(lane & 15) * stride + ((lane >> 4) << 3);
  v16h f;
#pragma unroll
  for (int i = 0; i < 8; ++i) { f[i] = p[i]; f[i + 8] = p[i + 16]; }
  return f;
}

// ---- pass 0a: x (fp32) -> xh (f16), one-time conversion --------------------
__global__ void __launch_bounds__(256)
cvt_x_f16(const float* __restrict__ x, _Float16* __restrict__ xh, int n) {
  const int i = (blockIdx.x * 256 + threadIdx.x) * 4;
  if (i + 3 < n) {
    const float4 v = *(const float4*)(x + i);
    xh[i]     = (_Float16)v.x;
    xh[i + 1] = (_Float16)v.y;
    xh[i + 2] = (_Float16)v.z;
    xh[i + 3] = (_Float16)v.w;
  }
}

// ---- pass 0b: Wq/Wk/Wv (fp32, row-major [k][col]) -> wt (f16, [col][k]) ----
__global__ void __launch_bounds__(256)
cvt_w_f16_t(const float* __restrict__ Wq, const float* __restrict__ Wk,
            const float* __restrict__ Wv, _Float16* __restrict__ wt) {
  const int t     = blockIdx.x * 256 + threadIdx.x;   // 0 .. 768*768-1
  const int which = blockIdx.y;
  const float* W  = (which == 0) ? Wq : (which == 1) ? Wk : Wv;
  const int col = t / kE;
  const int k   = t - col * kE;
  wt[(size_t)which * kE * kE + t] = (_Float16)W[(size_t)k * kE + col];
}

// ---- pass 1: QKV projection GEMM; one wave = 16x64 output tile -------------
__global__ void __launch_bounds__(32)
qkv_proj_wmma(const _Float16* __restrict__ xh, const _Float16* __restrict__ wt,
              const float* __restrict__ bq, const float* __restrict__ bk,
              const float* __restrict__ bv,
              _Float16* __restrict__ qh, _Float16* __restrict__ kh,
              _Float16* __restrict__ vt) {
  const int lane  = threadIdx.x;
  const int rt    = blockIdx.x;   // 16-row tile of B*N
  const int ct    = blockIdx.y;   // 64-col tile of E
  const int which = blockIdx.z;   // 0=Q 1=K 2=V

  const _Float16* Wt   = wt + (size_t)which * kE * kE;  // f16 [col][k]
  const float*    bias = (which == 0) ? bq : (which == 1) ? bk : bv;
  const _Float16* arow = xh + (size_t)rt * 16 * kE;

  v8f acc[4] = {};
  for (int kc = 0; kc < kE; kc += 32) {
    const v16h a = load_frag16(arow + kc, kE, lane);
    __builtin_prefetch(arow + kc + 32, 0, 1);
#pragma unroll
    for (int sub = 0; sub < 4; ++sub) {
      const v16h b =
          load_frag16(Wt + (size_t)(ct * 64 + sub * 16) * kE + kc, kE, lane);
      acc[sub] = wmma_f16(a, b, acc[sub]);
    }
  }

#pragma unroll
  for (int sub = 0; sub < 4; ++sub) {
    const int   bcol = ct * 64 + sub * 16 + (lane & 15);
    const float bval = bias[bcol];
    const int   hsel = bcol / kD;
    const int   d    = bcol - hsel * kD;
#pragma unroll
    for (int r = 0; r < 8; ++r) {
      const int R    = rt * 16 + r + ((lane >> 4) << 3);  // global row in B*N
      const int bidx = R / kN;
      const int n    = R - bidx * kN;
      const int bh   = bidx * kH + hsel;
      const _Float16 val = (_Float16)(acc[sub][r] + bval);
      if (which == 0)      qh[((size_t)bh * kNP + n) * kD + d] = val;
      else if (which == 1) kh[((size_t)bh * kNP + n) * kD + d] = val;
      else                 vt[((size_t)bh * kD + d) * kNP + n] = val;  // V^T
    }
  }
}

// ---- pass 2: per-(b,h,query-tile) attention: QK^T -> softmax -> PV ---------
__global__ void __launch_bounds__(32)
attn_wmma(const _Float16* __restrict__ qh, const _Float16* __restrict__ kh,
          const _Float16* __restrict__ vt, float* __restrict__ out) {
  const int lane = threadIdx.x;
  const int qb   = blockIdx.x * 16;   // query tile base
  const int bh   = blockIdx.y;        // b*8 + h

  __shared__ float    s[16][kNP + 16];      // energy (fp32), 16B-aligned rows
  __shared__ _Float16 pbuf[16][kNP + 16];   // exp probabilities (f16)
  __shared__ float    invs[16];             // scale / rowsum

  const _Float16* qbase = qh + ((size_t)bh * kNP + qb) * kD;
  const _Float16* kbase = kh + (size_t)bh * kNP * kD;
  const _Float16* vbase = vt + (size_t)bh * kD * kNP;

  // Q tile 16x96 -> 3 A fragments (K-dim = d, chunks of 32)
  v16h qa[3];
#pragma unroll
  for (int c = 0; c < 3; ++c) qa[c] = load_frag16(qbase + c * 32, kD, lane);

  // Phase A: energy tiles S = Q K^T, masked pad keys to -1e30, park in LDS
  for (int kb = 0; kb < kNP; kb += 16) {
    __builtin_prefetch(kbase + (size_t)(kb + 16) * kD + (lane << 3), 0, 1);
    v8f e = {};
#pragma unroll
    for (int c = 0; c < 3; ++c) {
      const v16h kf = load_frag16(kbase + (size_t)kb * kD + c * 32, kD, lane);
      e = wmma_f16(qa[c], kf, e);
    }
    const int  col   = lane & 15;
    const bool valid = (kb + col) < kN;
#pragma unroll
    for (int r = 0; r < 8; ++r) {
      const int row = r + ((lane >> 4) << 3);
      s[row][kb + col] = valid ? e[r] : -1e30f;
    }
  }
  __syncthreads();

  // Phase B: rowwise softmax; each half-wave covers half the key range
  {
    const int row = lane & 15;
    const int c0  = (lane >> 4) * (kNP / 2);
    float mx = -1e30f;
    for (int c = 0; c < kNP / 2; ++c) mx = fmaxf(mx, s[row][c0 + c]);
    mx = fmaxf(mx, __shfl_xor(mx, 16, 32));
    float sum = 0.f;
    for (int c = 0; c < kNP / 2; ++c) {
      const float ev = __expf(s[row][c0 + c] - mx);
      sum += ev;
      pbuf[row][c0 + c] = (_Float16)ev;   // pad cols get exact 0
    }
    sum += __shfl_xor(sum, 16, 32);
    if ((lane >> 4) == 0) invs[row] = kScale / sum;
  }
  __syncthreads();

  // Phase C: O = P @ V, 32 keys per step, 6 d-tiles of 16
  v8f o[6] = {};
  for (int kb = 0; kb < kNP; kb += 32) {
    __builtin_prefetch(vbase + (size_t)(lane * 16) * kNP + kb + 32, 0, 1);
    const v16h pa = load_frag16(&pbuf[0][kb], kNP + 16, lane);
#pragma unroll
    for (int dt = 0; dt < 6; ++dt) {
      const v16h vf = load_frag16(vbase + (size_t)(dt * 16) * kNP + kb, kNP, lane);
      o[dt] = wmma_f16(pa, vf, o[dt]);
    }
  }

  // Finalize: scale by (D^-0.5 / rowsum), scatter to [b, n, h*96 + d]
  const int b = bh >> 3, h = bh & 7;
  const int coln = lane & 15;
#pragma unroll
  for (int dt = 0; dt < 6; ++dt) {
#pragma unroll
    for (int r = 0; r < 8; ++r) {
      const int row  = r + ((lane >> 4) << 3);
      const int qrow = qb + row;
      if (qrow < kN)
        out[((size_t)b * kN + qrow) * kE + h * kD + dt * 16 + coln] =
            o[dt][r] * invs[row];
    }
  }
}

// ---- host entry ------------------------------------------------------------
extern "C" void kernel_launch(void* const* d_in, const int* in_sizes, int n_in,
                              void* d_out, int out_size, void* d_ws, size_t ws_size,
                              hipStream_t stream) {
  (void)in_sizes; (void)n_in; (void)out_size; (void)ws_size;
  const float* x  = (const float*)d_in[0];
  const float* Wq = (const float*)d_in[1];
  const float* bq = (const float*)d_in[2];
  const float* Wk = (const float*)d_in[3];
  const float* bk = (const float*)d_in[4];
  const float* Wv = (const float*)d_in[5];
  const float* bv = (const float*)d_in[6];
  float* out = (float*)d_out;

  // workspace layout (all f16)
  const size_t nXh    = (size_t)kBN * kE;           // 28,360,704
  const size_t nWt    = (size_t)3 * kE * kE;        //  1,769,472
  const size_t perBuf = (size_t)kB * kH * kNP * kD; // 29,884,416 each
  _Float16* xh = (_Float16*)d_ws;
  _Float16* wt = xh + nXh;
  _Float16* qh = wt + nWt;
  _Float16* kh = qh + perBuf;
  _Float16* vt = kh + perBuf;

  // zero the Q/K/V staging (covers the N->608 padding); graph-capturable
  hipMemsetAsync(qh, 0, 3 * perBuf * sizeof(_Float16), stream);

  // pass 0: one-time fp32 -> f16 conversions (W transposed to col-major)
  cvt_x_f16<<<dim3((unsigned)((nXh / 4 + 255) / 256)), 256, 0, stream>>>(
      x, xh, (int)nXh);
  cvt_w_f16_t<<<dim3(kE * kE / 256, 3), 256, 0, stream>>>(Wq, Wk, Wv, wt);

  // pass 1: QKV projection (16x64 tile per wave)
  dim3 gp(kBN / 16, kE / 64, 3);
  qkv_proj_wmma<<<gp, 32, 0, stream>>>(xh, wt, bq, bk, bv, qh, kh, vt);

  // pass 2: attention
  dim3 ga((kN + 15) / 16, kB * kH);
  attn_wmma<<<ga, 32, 0, stream>>>(qh, kh, vt, out);
}
